// TransformerLayer_30880814859142
// MI455X (gfx1250) — matrix operand order repaired and checked by reading, not compiled, not run on hardware
//
#include <hip/hip_runtime.h>
#include <hip/hip_bf16.h>

typedef __attribute__((ext_vector_type(16))) _Float16 v16h;
typedef __attribute__((ext_vector_type(8)))  _Float16 v8h;
typedef __attribute__((ext_vector_type(4)))  _Float16 v4h;
typedef __attribute__((ext_vector_type(8)))  float    v8f;

#define E_DIM 256
#define H_NUM 8
#define D_DIM 32
#define F_DIM 1024
#define SEQ   2048
#define BATCH 4
#define MROWS (BATCH * SEQ)   /* 8192 */

// ---------------------------------------------------------------------------
// WMMA helpers (CDNA5 wave32, V_WMMA_F32_16X16X32_F16)
// ---------------------------------------------------------------------------
__device__ __forceinline__ v8f wmma16(v16h a, v16h b, v8f c) {
  return __builtin_amdgcn_wmma_f32_16x16x32_f16(false, a, false, b, (short)0, c,
                                                false, false);
}

// A-fragment (16x32 f16, row-major source, stride lda).
// Lane layout: row m = lane&15; a[j] = A[m][k0 + hi*16 + j] (contiguous 16).
__device__ __forceinline__ v16h load_afrag(const _Float16* A, int lda, int m0, int k0) {
  int lane = threadIdx.x & 31;
  const _Float16* p = A + (size_t)(m0 + (lane & 15)) * lda + k0 + ((lane >> 4) << 4);
  v8h lo = *(const v8h*)p;
  v8h hi = *(const v8h*)(p + 8);
  v16h r;
#pragma unroll
  for (int i = 0; i < 8; ++i) { r[i] = lo[i]; r[i + 8] = hi[i]; }
  return r;
}

// B-fragment for B = W^T where W is [N,K] row-major (weights / K-rows / V^T-rows).
// Lane layout: col n = lane&15; b[j] = W[n][k0 + hi*16 + j], one contiguous v16h.
__device__ __forceinline__ v16h load_bfrag(const _Float16* W, int ldw, int n0, int k0) {
  int lane = threadIdx.x & 31;
  return *(const v16h*)(W + (size_t)(n0 + (lane & 15)) * ldw + k0 + ((lane >> 4) << 4));
}

// ---------------------------------------------------------------------------
// f32 -> f16 weight conversion
// ---------------------------------------------------------------------------
__global__ void f32_to_f16_kernel(const float* __restrict__ in,
                                  _Float16* __restrict__ out, int n) {
  int i = blockIdx.x * blockDim.x + threadIdx.x;
  if (i < n) out[i] = (_Float16)in[i];
}

// ---------------------------------------------------------------------------
// LayerNorm (one wave per 256-element row) with f16 output
// ---------------------------------------------------------------------------
__global__ void __launch_bounds__(256)
layernorm_f16_kernel(const float* __restrict__ x, const float* __restrict__ g,
                     const float* __restrict__ beta, _Float16* __restrict__ out) {
  int wave = (blockIdx.x * blockDim.x + threadIdx.x) >> 5;
  int lane = threadIdx.x & 31;
  const float4* row = (const float4*)(x + (size_t)wave * E_DIM);
  float4 a = row[lane];
  float4 b = row[32 + lane];
  float s = a.x + a.y + a.z + a.w + b.x + b.y + b.z + b.w;
  float q = a.x * a.x + a.y * a.y + a.z * a.z + a.w * a.w +
            b.x * b.x + b.y * b.y + b.z * b.z + b.w * b.w;
#pragma unroll
  for (int off = 16; off > 0; off >>= 1) {
    s += __shfl_xor(s, off, 32);
    q += __shfl_xor(q, off, 32);
  }
  float mu = s * (1.0f / E_DIM);
  float var = q * (1.0f / E_DIM) - mu * mu;
  float rs = rsqrtf(var + 1e-5f);
  const float4* g4 = (const float4*)g;
  const float4* b4 = (const float4*)beta;
  float4 ga = g4[lane], gb = g4[32 + lane];
  float4 ba = b4[lane], bb = b4[32 + lane];
  _Float16* o = out + (size_t)wave * E_DIM;
  v4h o0, o1;
  o0[0] = (_Float16)((a.x - mu) * rs * ga.x + ba.x);
  o0[1] = (_Float16)((a.y - mu) * rs * ga.y + ba.y);
  o0[2] = (_Float16)((a.z - mu) * rs * ga.z + ba.z);
  o0[3] = (_Float16)((a.w - mu) * rs * ga.w + ba.w);
  o1[0] = (_Float16)((b.x - mu) * rs * gb.x + bb.x);
  o1[1] = (_Float16)((b.y - mu) * rs * gb.y + bb.y);
  o1[2] = (_Float16)((b.z - mu) * rs * gb.z + bb.z);
  o1[3] = (_Float16)((b.w - mu) * rs * gb.w + bb.w);
  *(v4h*)(o + lane * 4) = o0;
  *(v4h*)(o + 128 + lane * 4) = o1;
}

// ---------------------------------------------------------------------------
// Generic WMMA GEMM: C[M,N] = A[M,K] @ W[N,K]^T  (+bias, relu, +residual)
// One wave computes a 32x32 tile (2 A-frags x 2 B-frags x 4 accum per K step).
// ---------------------------------------------------------------------------
template <int RELU, int BIAS, int RES, int OUTF, int OUTH>
__global__ void __launch_bounds__(256)
gemm_kernel(const _Float16* __restrict__ A, const _Float16* __restrict__ W,
            const float* __restrict__ bias, const float* __restrict__ res,
            float* __restrict__ outF, _Float16* __restrict__ outH,
            int M, int N, int K) {
  int wave = blockIdx.x * (blockDim.x >> 5) + (threadIdx.x >> 5);
  int lane = threadIdx.x & 31;
  int tilesN = N >> 5;
  int tm = (wave / tilesN) << 5;
  int tn = (wave % tilesN) << 5;
  if (tm >= M) return;
  v8f c00 = {}, c01 = {}, c10 = {}, c11 = {};
  for (int k = 0; k < K; k += 32) {
    v16h a0 = load_afrag(A, K, tm, k);
    v16h a1 = load_afrag(A, K, tm + 16, k);
    v16h b0 = load_bfrag(W, K, tn, k);
    v16h b1 = load_bfrag(W, K, tn + 16, k);
    c00 = wmma16(a0, b0, c00);
    c01 = wmma16(a0, b1, c01);
    c10 = wmma16(a1, b0, c10);
    c11 = wmma16(a1, b1, c11);
  }
  int cn = lane & 15, hi = lane >> 4;
#pragma unroll
  for (int t = 0; t < 4; ++t) {
    v8f c = (t == 0) ? c00 : (t == 1) ? c01 : (t == 2) ? c10 : c11;
    int mi = (t >> 1) << 4, ni = (t & 1) << 4;
#pragma unroll
    for (int v = 0; v < 8; ++v) {
      int gm = tm + mi + (hi << 3) + v;
      int gn = tn + ni + cn;
      float val = c[v];
      if (BIAS) val += bias[gn];
      if (RELU) val = fmaxf(val, 0.0f);
      if (RES) val += res[(size_t)gm * N + gn];
      if (OUTF) outF[(size_t)gm * N + gn] = val;
      if (OUTH) outH[(size_t)gm * N + gn] = (_Float16)val;
    }
  }
}

// ---------------------------------------------------------------------------
// QKV GEMM: [8192,256] @ w_in[768,256]^T, scattering into
//   Q [B,H,S,D], K [B,H,S,D], V^T [B,H,D,S]  (all f16)
// ---------------------------------------------------------------------------
__global__ void __launch_bounds__(256)
qkv_gemm_kernel(const _Float16* __restrict__ A, const _Float16* __restrict__ W,
                _Float16* __restrict__ Qo, _Float16* __restrict__ Ko,
                _Float16* __restrict__ Vt) {
  const int Kd = E_DIM, N = 3 * E_DIM;
  int wave = blockIdx.x * (blockDim.x >> 5) + (threadIdx.x >> 5);
  int lane = threadIdx.x & 31;
  int tilesN = N >> 5;  // 24
  int tm = (wave / tilesN) << 5;
  int tn = (wave % tilesN) << 5;
  if (tm >= MROWS) return;
  v8f c00 = {}, c01 = {}, c10 = {}, c11 = {};
  for (int k = 0; k < Kd; k += 32) {
    v16h a0 = load_afrag(A, Kd, tm, k);
    v16h a1 = load_afrag(A, Kd, tm + 16, k);
    v16h b0 = load_bfrag(W, Kd, tn, k);
    v16h b1 = load_bfrag(W, Kd, tn + 16, k);
    c00 = wmma16(a0, b0, c00);
    c01 = wmma16(a0, b1, c01);
    c10 = wmma16(a1, b0, c10);
    c11 = wmma16(a1, b1, c11);
  }
  int cn = lane & 15, hi = lane >> 4;
#pragma unroll
  for (int t = 0; t < 4; ++t) {
    v8f c = (t == 0) ? c00 : (t == 1) ? c01 : (t == 2) ? c10 : c11;
    int mi = (t >> 1) << 4, ni = (t & 1) << 4;
#pragma unroll
    for (int v = 0; v < 8; ++v) {
      int gm = tm + mi + (hi << 3) + v;
      int gn = tn + ni + cn;
      int b = gm >> 11, s = gm & (SEQ - 1);
      int reg = gn >> 8;       // 0=q, 1=k, 2=v
      int n2 = gn & 255;
      int h = n2 >> 5, d = n2 & 31;
      _Float16 hv = (_Float16)c[v];
      size_t bh = (size_t)(b * H_NUM + h);
      if (reg == 0)
        Qo[(bh * SEQ + s) * D_DIM + d] = hv;
      else if (reg == 1)
        Ko[(bh * SEQ + s) * D_DIM + d] = hv;
      else
        Vt[(bh * D_DIM + d) * SEQ + s] = hv;
    }
  }
}

// ---------------------------------------------------------------------------
// Flash attention: one wave = 16 query rows of one (b,h); online softmax,
// 32-key chunks: 2 WMMA for Q@K^T, 2 WMMA for P@V. P transposed via LDS.
// ---------------------------------------------------------------------------
__global__ void __launch_bounds__(128)
attention_kernel(const _Float16* __restrict__ Q, const _Float16* __restrict__ Kb,
                 const _Float16* __restrict__ Vt, _Float16* __restrict__ Oh) {
  __shared__ __align__(32) _Float16 pls[4][16 * 32];
  int wv = threadIdx.x >> 5;
  int lane = threadIdx.x & 31;
  int cn = lane & 15, hi = lane >> 4;
  int gw = blockIdx.x * 4 + wv;
  int qt = gw & 127;  // 128 q-tiles per (b,h)
  int bh = gw >> 7;   // 0..31
  const _Float16* Qh = Q + (size_t)bh * SEQ * D_DIM;
  const _Float16* Kh = Kb + (size_t)bh * SEQ * D_DIM;
  const _Float16* Vh = Vt + (size_t)bh * D_DIM * SEQ;
  int m0 = qt << 4;
  v16h qa = load_afrag(Qh, D_DIM, m0, 0);
  const float scale = 0.17677669529663687f;  // 1/sqrt(32)
  float mrun[8], lrun[8];
  v8f o0 = {}, o1 = {};
#pragma unroll
  for (int v = 0; v < 8; ++v) { mrun[v] = -3.0e38f; lrun[v] = 0.0f; }
  _Float16* pw = pls[wv];
  for (int kc = 0; kc < SEQ; kc += 32) {
    v16h bk0 = load_bfrag(Kh, D_DIM, kc, 0);
    v16h bk1 = load_bfrag(Kh, D_DIM, kc + 16, 0);
    v8f z = {};
    v8f s0 = wmma16(qa, bk0, z);
    v8f s1 = wmma16(qa, bk1, z);
    float p0[8], p1[8];
#pragma unroll
    for (int v = 0; v < 8; ++v) {
      float sa = s0[v] * scale, sb = s1[v] * scale;
      float mx = fmaxf(sa, sb);
      mx = fmaxf(mx, __shfl_xor(mx, 1, 32));
      mx = fmaxf(mx, __shfl_xor(mx, 2, 32));
      mx = fmaxf(mx, __shfl_xor(mx, 4, 32));
      mx = fmaxf(mx, __shfl_xor(mx, 8, 32));
      float mnew = fmaxf(mrun[v], mx);
      float corr = __expf(mrun[v] - mnew);
      p0[v] = __expf(sa - mnew);
      p1[v] = __expf(sb - mnew);
      float rs = p0[v] + p1[v];
      rs += __shfl_xor(rs, 1, 32);
      rs += __shfl_xor(rs, 2, 32);
      rs += __shfl_xor(rs, 4, 32);
      rs += __shfl_xor(rs, 8, 32);
      lrun[v] = lrun[v] * corr + rs;
      mrun[v] = mnew;
      o0[v] *= corr;
      o1[v] *= corr;
    }
    // C-layout -> A-layout transpose of P through per-wave LDS slice
#pragma unroll
    for (int v = 0; v < 8; ++v) {
      pw[((hi << 3) + v) * 32 + cn] = (_Float16)p0[v];
      pw[((hi << 3) + v) * 32 + 16 + cn] = (_Float16)p1[v];
    }
    v16h pa = load_afrag(pw, 32, 0, 0);
    v16h bv0 = *(const v16h*)(Vh + (size_t)cn * SEQ + kc + (hi << 4));
    v16h bv1 = *(const v16h*)(Vh + (size_t)(16 + cn) * SEQ + kc + (hi << 4));
    o0 = wmma16(pa, bv0, o0);
    o1 = wmma16(pa, bv1, o1);
  }
  int b = bh >> 3, h = bh & 7;
  int gmbase = b * SEQ + m0;
#pragma unroll
  for (int v = 0; v < 8; ++v) {
    float inv = 1.0f / lrun[v];
    int gm = gmbase + (hi << 3) + v;
    int col = h * 32 + cn;
    Oh[(size_t)gm * E_DIM + col] = (_Float16)(o0[v] * inv);
    Oh[(size_t)gm * E_DIM + col + 16] = (_Float16)(o1[v] * inv);
  }
}

// ---------------------------------------------------------------------------
// Host-side orchestration
// ---------------------------------------------------------------------------
extern "C" void kernel_launch(void* const* d_in, const int* in_sizes, int n_in,
                              void* d_out, int out_size, void* d_ws, size_t ws_size,
                              hipStream_t stream) {
  const float* src   = (const float*)d_in[0];
  const float* w_in  = (const float*)d_in[1];
  const float* w_out = (const float*)d_in[2];
  const float* w1    = (const float*)d_in[3];
  const float* b1    = (const float*)d_in[4];
  const float* w2    = (const float*)d_in[5];
  const float* b2    = (const float*)d_in[6];
  const float* g1    = (const float*)d_in[7];
  const float* beta1 = (const float*)d_in[8];
  const float* g2    = (const float*)d_in[9];
  const float* beta2 = (const float*)d_in[10];
  float* out = (float*)d_out;

  // Workspace carve-up
  char* p = (char*)d_ws;
  _Float16* xh     = (_Float16*)p; p += (size_t)MROWS * E_DIM * 2;       // 4 MB
  _Float16* qb     = (_Float16*)p; p += (size_t)MROWS * E_DIM * 2;       // 4 MB
  _Float16* kb     = (_Float16*)p; p += (size_t)MROWS * E_DIM * 2;       // 4 MB
  _Float16* vt     = (_Float16*)p; p += (size_t)MROWS * E_DIM * 2;       // 4 MB
  _Float16* attn_h = (_Float16*)p; p += (size_t)MROWS * E_DIM * 2;       // 4 MB
  float*    src2   = (float*)p;    p += (size_t)MROWS * E_DIM * 4;       // 8 MB
  _Float16* yh     = (_Float16*)p; p += (size_t)MROWS * E_DIM * 2;       // 4 MB
  _Float16* y1h    = (_Float16*)p; p += (size_t)MROWS * F_DIM * 2;       // 16 MB
  _Float16* w_in_h  = (_Float16*)p; p += (size_t)3 * E_DIM * E_DIM * 2;
  _Float16* w_out_h = (_Float16*)p; p += (size_t)E_DIM * E_DIM * 2;
  _Float16* w1_h    = (_Float16*)p; p += (size_t)F_DIM * E_DIM * 2;
  _Float16* w2_h    = (_Float16*)p; p += (size_t)E_DIM * F_DIM * 2;

  // 1) Convert weights to f16
  f32_to_f16_kernel<<<(3 * E_DIM * E_DIM) / 256, 256, 0, stream>>>(w_in, w_in_h, 3 * E_DIM * E_DIM);
  f32_to_f16_kernel<<<(E_DIM * E_DIM) / 256, 256, 0, stream>>>(w_out, w_out_h, E_DIM * E_DIM);
  f32_to_f16_kernel<<<(F_DIM * E_DIM) / 256, 256, 0, stream>>>(w1, w1_h, F_DIM * E_DIM);
  f32_to_f16_kernel<<<(E_DIM * F_DIM) / 256, 256, 0, stream>>>(w2, w2_h, E_DIM * F_DIM);

  // 2) LN1 -> f16
  layernorm_f16_kernel<<<MROWS / 8, 256, 0, stream>>>(src, g1, beta1, xh);

  // 3) QKV GEMM with scatter to Q/K/V^T layouts
  {
    int tiles = (MROWS / 32) * ((3 * E_DIM) / 32);  // 6144
    qkv_gemm_kernel<<<tiles / 8, 256, 0, stream>>>(xh, w_in_h, qb, kb, vt);
  }

  // 4) Flash attention
  attention_kernel<<<(BATCH * H_NUM * (SEQ / 16)) / 4, 128, 0, stream>>>(qb, kb, vt, attn_h);

  // 5) Out-projection + residual: src2 = src + attn @ w_out^T
  {
    int tiles = (MROWS / 32) * (E_DIM / 32);  // 2048
    gemm_kernel<0, 0, 1, 1, 0><<<tiles / 8, 256, 0, stream>>>(
        attn_h, w_out_h, nullptr, src, src2, nullptr, MROWS, E_DIM, E_DIM);
  }

  // 6) LN2 -> f16
  layernorm_f16_kernel<<<MROWS / 8, 256, 0, stream>>>(src2, g2, beta2, yh);

  // 7) FFN1: relu(yh @ w1^T + b1) -> f16
  {
    int tiles = (MROWS / 32) * (F_DIM / 32);  // 8192
    gemm_kernel<1, 1, 0, 0, 1><<<tiles / 8, 256, 0, stream>>>(
        yh, w1_h, b1, nullptr, nullptr, y1h, MROWS, F_DIM, E_DIM);
  }

  // 8) FFN2: out = src2 + y1 @ w2^T + b2
  {
    int tiles = (MROWS / 32) * (E_DIM / 32);  // 2048
    gemm_kernel<0, 1, 1, 1, 0><<<tiles / 8, 256, 0, stream>>>(
        y1h, w2_h, b2, src2, out, nullptr, MROWS, E_DIM, F_DIM);
  }
}
